// EncoderLayer_68917045231712
// MI455X (gfx1250) — matrix operand order repaired and checked by reading, not compiled, not run on hardware
//
#include <hip/hip_runtime.h>
#include <hip/hip_bf16.h>
#include <math.h>

// ---------------------------------------------------------------------------
// Problem constants (match reference)
// ---------------------------------------------------------------------------
#define Lnodes 20000
#define Eedges 320000
#define Cdim   256
#define Hheads 8
#define DHdim  32
#define HIDdim 512
#define SCALE_ATT 0.17677669529663689f  // 1/sqrt(32)

typedef __bf16 bf16_t;
typedef __attribute__((ext_vector_type(16))) __bf16 v16bf;
typedef __attribute__((ext_vector_type(8)))  float  v8f;

// ---------------------------------------------------------------------------
// Small helpers
// ---------------------------------------------------------------------------
__device__ __forceinline__ float gelu_tanh(float x) {
    // jax.nn.gelu default (approximate=True)
    float x3 = x * x * x;
    float t  = tanhf(0.7978845608028654f * (x + 0.044715f * x3));
    return 0.5f * x * (1.0f + t);
}

__device__ __forceinline__ void atomicMaxF(float* addr, float val) {
    int old = __float_as_int(*addr);
    while (__int_as_float(old) < val) {
        int assumed = old;
        old = atomicCAS((int*)addr, assumed, __float_as_int(val));
        if (old == assumed) break;
    }
}

// ---------------------------------------------------------------------------
// Utility kernels
// ---------------------------------------------------------------------------
__global__ void k_fill_f32(float* p, float v, int n) {
    int i = blockIdx.x * blockDim.x + threadIdx.x;
    if (i < n) p[i] = v;
}

// Convert an f32 weight matrix W[K,N] (row major) into bf16 packed in WMMA
// B-fragment order: for tile (tn, tk) of shape K32 x N16, lane l owns
// K = tk*32 + (l>>4)*16 + i, N = tn*16 + (l&15), i = 0..15 stored contiguously,
// so a lane's whole fragment is one aligned 32-byte v16bf load.
__global__ void k_pack_b_bf16(const float* __restrict__ W, bf16_t* __restrict__ P,
                              int K, int N) {
    int idx = blockIdx.x * blockDim.x + threadIdx.x;
    if (idx >= K * N) return;
    int i    = idx & 15;
    int lane = (idx >> 4) & 31;
    int t    = idx >> 9;
    int KT   = K >> 5;
    int tk   = t % KT;
    int tn   = t / KT;
    int k = tk * 32 + (lane >> 4) * 16 + i;
    int n = tn * 16 + (lane & 15);
    P[idx] = (bf16_t)W[(size_t)k * N + n];
}

// ---------------------------------------------------------------------------
// LayerNorm: one wave per node, 8 channels per lane (C=256, wave32)
// ---------------------------------------------------------------------------
__global__ void k_layernorm(const float* __restrict__ x,
                            const float* __restrict__ g,
                            const float* __restrict__ b,
                            float* __restrict__ z,
                            float* __restrict__ xcopy) {
    int node = blockIdx.x * 8 + (threadIdx.x >> 5);
    int lane = threadIdx.x & 31;
    if (node >= Lnodes) return;
    const float* xr = x + (size_t)node * Cdim;

    float vals[8];
    float s = 0.f;
    #pragma unroll
    for (int i = 0; i < 8; ++i) { vals[i] = xr[lane + 32 * i]; s += vals[i]; }
    #pragma unroll
    for (int m = 16; m >= 1; m >>= 1) s += __shfl_xor(s, m, 32);
    float mu = s * (1.0f / Cdim);

    float vs = 0.f;
    #pragma unroll
    for (int i = 0; i < 8; ++i) { float d0 = vals[i] - mu; vs += d0 * d0; }
    #pragma unroll
    for (int m = 16; m >= 1; m >>= 1) vs += __shfl_xor(vs, m, 32);
    float rstd = rsqrtf(vs * (1.0f / Cdim) + 1e-5f);

    float* zr = z + (size_t)node * Cdim;
    #pragma unroll
    for (int i = 0; i < 8; ++i) {
        int c = lane + 32 * i;
        zr[c] = (vals[i] - mu) * rstd * g[c] + b[c];
        if (xcopy) xcopy[(size_t)node * Cdim + c] = vals[i];
    }
}

// ---------------------------------------------------------------------------
// WMMA GEMM, 16x64 output tile per wave (4 accumulators), K templated so the
// K-loop fully unrolls into back-to-back v_wmma ops.
//   out[M,N] = epilogue( A[M,K] @ B[K,N] )
//   A: fp32, converted to bf16 in the fragment loader (v_cvt_pk_bf16_f32),
//      optional row gather (A[Aidx[m]]).
//   Bp: bf16 weights pre-packed in fragment-major order (k_pack_b_bf16).
//   Epilogue: +bias[n], optional tanh-GELU, + up to two fp32 residuals.
// Block = 128 threads = 4 waves; wave w covers columns (blockIdx.y*4+w)*64.
// grid = (M/16, N/256). Requires M%16==0, N%256==0, K%32==0 (true here).
// ---------------------------------------------------------------------------
template<int K, bool GATHER>
__global__ void k_gemm_wmma(const float* __restrict__ A,
                            const int*   __restrict__ Aidx,
                            const bf16_t* __restrict__ Bp,
                            const float* __restrict__ bias,
                            const float* __restrict__ res1,
                            const float* __restrict__ res2,
                            float* __restrict__ out,
                            int N, int act) {
    constexpr int KT = K / 32;
    const int lane  = threadIdx.x & 31;
    const int wave  = threadIdx.x >> 5;
    const int bm    = blockIdx.x * 16;
    const int nbase = (blockIdx.y * 4 + wave) * 64;
    const int tn0   = nbase >> 4;          // first of 4 N-tiles for this wave

    int arow = bm + (lane & 15);
    if (GATHER) arow = Aidx[arow];
    const float* Ar = A + (size_t)arow * K;
    const int aoff  = (lane >> 4) * 16;

    const v16bf* B = (const v16bf*)Bp;
    const size_t bbase = ((size_t)tn0 * KT) * 32 + lane;

    v8f acc0 = {}, acc1 = {}, acc2 = {}, acc3 = {};
    #pragma unroll
    for (int t = 0; t < KT; ++t) {
        const int k0 = t * 32;
        v16bf af;
        #pragma unroll
        for (int i = 0; i < 8; ++i) {
            af[i]     = (bf16_t)Ar[k0 + aoff + i];        // K = 0..7  / 16..23
            af[8 + i] = (bf16_t)Ar[k0 + aoff + 8 + i];    // K = 8..15 / 24..31
        }
        v16bf b0 = B[bbase + (size_t)(0 * KT + t) * 32];
        v16bf b1 = B[bbase + (size_t)(1 * KT + t) * 32];
        v16bf b2 = B[bbase + (size_t)(2 * KT + t) * 32];
        v16bf b3 = B[bbase + (size_t)(3 * KT + t) * 32];
        acc0 = __builtin_amdgcn_wmma_f32_16x16x32_bf16(false, af, false, b0, (short)0, acc0, false, false);
        acc1 = __builtin_amdgcn_wmma_f32_16x16x32_bf16(false, af, false, b1, (short)0, acc1, false, false);
        acc2 = __builtin_amdgcn_wmma_f32_16x16x32_bf16(false, af, false, b2, (short)0, acc2, false, false);
        acc3 = __builtin_amdgcn_wmma_f32_16x16x32_bf16(false, af, false, b3, (short)0, acc3, false, false);
    }

    const int mrow0 = (lane >> 4) * 8;
    const int nloc  = lane & 15;
    #pragma unroll
    for (int j = 0; j < 4; ++j) {
        v8f acc = (j == 0) ? acc0 : (j == 1) ? acc1 : (j == 2) ? acc2 : acc3;
        int n = nbase + j * 16 + nloc;
        float bv = bias ? bias[n] : 0.f;
        #pragma unroll
        for (int i = 0; i < 8; ++i) {
            int gr = bm + mrow0 + i;
            size_t o = (size_t)gr * N + n;
            float vl = acc[i] + bv;
            if (act)  vl = gelu_tanh(vl);
            if (res1) vl += res1[o];
            if (res2) vl += res2[o];
            out[o] = vl;
        }
    }
}

// ---------------------------------------------------------------------------
// Edge kernels: thread per (edge, head)
// ---------------------------------------------------------------------------
__global__ void k_edge_scores(const float* __restrict__ q,
                              const float* __restrict__ k,
                              const int* __restrict__ rows,
                              const int* __restrict__ cols,
                              const float* __restrict__ bias,
                              float* __restrict__ scores,
                              float* __restrict__ rowmax) {
    int idx = blockIdx.x * blockDim.x + threadIdx.x;
    if (idx >= Eedges * Hheads) return;
    int e = idx >> 3, h = idx & 7;
    int r = rows[e], c = cols[e];
    const float4* qp = (const float4*)(q + (size_t)r * Cdim + h * DHdim);
    const float4* kp = (const float4*)(k + (size_t)c * Cdim + h * DHdim);
    float s = 0.f;
    #pragma unroll
    for (int i = 0; i < 8; ++i) {
        float4 a = qp[i], b = kp[i];
        s += a.x * b.x + a.y * b.y + a.z * b.z + a.w * b.w;
    }
    s = s * SCALE_ATT + bias[idx];
    scores[idx] = s;
    atomicMaxF(&rowmax[r * Hheads + h], s);
}

__global__ void k_edge_expsum(float* __restrict__ scores,
                              const int* __restrict__ rows,
                              const float* __restrict__ rowmax,
                              float* __restrict__ rowsum) {
    int idx = blockIdx.x * blockDim.x + threadIdx.x;
    if (idx >= Eedges * Hheads) return;
    int e = idx >> 3, h = idx & 7;
    int r = rows[e];
    float ex = expf(scores[idx] - rowmax[r * Hheads + h]);
    scores[idx] = ex;
    atomicAdd(&rowsum[r * Hheads + h], ex);
}

__global__ void k_edge_msg(const float* __restrict__ scores,
                           const int* __restrict__ rows,
                           const int* __restrict__ cols,
                           const float* __restrict__ rowsum,
                           const float* __restrict__ v,
                           float* __restrict__ agg) {
    int idx = blockIdx.x * blockDim.x + threadIdx.x;
    if (idx >= Eedges * Hheads) return;
    int e = idx >> 3, h = idx & 7;
    int r = rows[e], c = cols[e];
    float alpha = scores[idx] / (rowsum[r * Hheads + h] + 1e-9f);
    const float* vv = v + (size_t)c * Cdim + h * DHdim;
    float* o = agg + (size_t)r * Cdim + h * DHdim;
    #pragma unroll
    for (int d = 0; d < DHdim; ++d) atomicAdd(&o[d], alpha * vv[d]);
}

__global__ void k_pos_msg(const float* __restrict__ scores,
                          const int* __restrict__ rows,
                          const int* __restrict__ srcs,
                          const float* __restrict__ rowsum,
                          const float* __restrict__ dist,
                          const float* __restrict__ pos,
                          const float* __restrict__ src_pos,
                          float* __restrict__ agg3) {
    int idx = blockIdx.x * blockDim.x + threadIdx.x;
    if (idx >= Eedges * Hheads) return;
    int e = idx >> 3, h = idx & 7;
    int r = rows[e], s = srcs[e];
    float alpha = scores[idx] / (rowsum[r * Hheads + h] + 1e-9f);
    float inv = 1.0f / dist[e];
    #pragma unroll
    for (int j = 0; j < 3; ++j) {
        float dirj = (src_pos[s * 3 + j] - pos[r * 3 + j]) * inv;
        atomicAdd(&agg3[r * (3 * Hheads) + h * 3 + j], alpha * dirj);
    }
}

// (L,24) @ (24,256) + bo, fused residual into xcur (K=24 too small for WMMA)
__global__ void k_pos_wo(const float* __restrict__ agg3,
                         const float* __restrict__ wo,
                         const float* __restrict__ bo,
                         float* __restrict__ xcur) {
    int idx = blockIdx.x * blockDim.x + threadIdx.x;
    if (idx >= Lnodes * Cdim) return;
    int l = idx >> 8, c = idx & 255;
    const float* a = agg3 + l * (3 * Hheads);
    float acc = bo[c];
    #pragma unroll
    for (int j = 0; j < 3 * Hheads; ++j) acc += a[j] * wo[j * Cdim + c];
    xcur[idx] += acc;
}

// ---------------------------------------------------------------------------
// Host-side orchestration
// ---------------------------------------------------------------------------
extern "C" void kernel_launch(void* const* d_in, const int* in_sizes, int n_in,
                              void* d_out, int out_size, void* d_ws, size_t ws_size,
                              hipStream_t stream) {
    (void)in_sizes; (void)n_in; (void)out_size; (void)ws_size;

    // ---- inputs (setup_inputs order) ----
    const float* x_in     = (const float*)d_in[0];
    const int*   row_idx  = (const int*)  d_in[1];
    const int*   col_idx  = (const int*)  d_in[2];
    const int*   src_idx  = (const int*)  d_in[3];
    const float* att_bias = (const float*)d_in[4];
    const float* pos_bias = (const float*)d_in[5];
    const float* dist     = (const float*)d_in[6];
    const float* pos      = (const float*)d_in[7];
    const float* src_pos  = (const float*)d_in[8];
    const int*   o2s      = (const int*)  d_in[9];
    const float* ln_att_g = (const float*)d_in[10];
    const float* ln_att_b = (const float*)d_in[11];
    const float* ln_pos_g = (const float*)d_in[12];
    const float* ln_pos_b = (const float*)d_in[13];
    const float* ln_ff_g  = (const float*)d_in[14];
    const float* ln_ff_b  = (const float*)d_in[15];
    const float* att_wq   = (const float*)d_in[16];
    const float* att_bq   = (const float*)d_in[17];
    const float* att_wk   = (const float*)d_in[18];
    const float* att_bk   = (const float*)d_in[19];
    const float* att_wv   = (const float*)d_in[20];
    const float* att_bv   = (const float*)d_in[21];
    const float* att_wo   = (const float*)d_in[22];
    const float* att_bo   = (const float*)d_in[23];
    const float* pos_wq   = (const float*)d_in[24];
    const float* pos_wk   = (const float*)d_in[25];
    const float* pos_wo   = (const float*)d_in[26];
    const float* pos_bo   = (const float*)d_in[27];
    const float* ff_w1    = (const float*)d_in[28];
    const float* ff_b1    = (const float*)d_in[29];
    const float* ff_w2    = (const float*)d_in[30];
    const float* ff_b2    = (const float*)d_in[31];

    // ---- workspace layout (floats) ----
    const size_t LC = (size_t)Lnodes * Cdim;          // 5,120,000
    const size_t EH = (size_t)Eedges * Hheads;        // 2,560,000
    const size_t LH = (size_t)Lnodes * Hheads;        //   160,000
    float* ws     = (float*)d_ws;
    float* xcur   = ws;                 // L x C running residual stream
    float* z      = ws + LC;            // L x C LayerNorm output
    float* q      = ws + 2 * LC;        // L x C
    float* k      = ws + 3 * LC;        // L x C   (q..k also aliases L x 512 hidden)
    float* v      = ws + 4 * LC;        // L x C
    float* hidden = q;                  // L x HID (spans q and k regions)
    float* scores = ws + 5 * LC;        // E x H
    float* rowmax = scores + EH;        // L x H
    float* rowsum = rowmax + LH;        // L x H
    float* agg    = rowsum + LH;        // L x C (attn agg; first L*24 reused for pos agg)
    bf16_t* bpool = (bf16_t*)(agg + LC);  // 32B-aligned (offset multiple of 32)
    const size_t CC = (size_t)Cdim * Cdim;
    bf16_t* wq_a = bpool;
    bf16_t* wk_a = bpool + CC;
    bf16_t* wv_a = bpool + 2 * CC;
    bf16_t* wo_a = bpool + 3 * CC;
    bf16_t* wq_p = bpool + 4 * CC;
    bf16_t* wk_p = bpool + 5 * CC;
    bf16_t* w1_b = bpool + 6 * CC;                       // C x HID packed
    bf16_t* w2_b = w1_b + (size_t)Cdim * HIDdim;         // HID x C packed

    const int T = 256;
    #define GRID1(n) dim3(((n) + T - 1) / T)

    // ---- weights fp32 -> bf16, packed into WMMA fragment-major order ----
    k_pack_b_bf16<<<GRID1(CC), T, 0, stream>>>(att_wq, wq_a, Cdim, Cdim);
    k_pack_b_bf16<<<GRID1(CC), T, 0, stream>>>(att_wk, wk_a, Cdim, Cdim);
    k_pack_b_bf16<<<GRID1(CC), T, 0, stream>>>(att_wv, wv_a, Cdim, Cdim);
    k_pack_b_bf16<<<GRID1(CC), T, 0, stream>>>(att_wo, wo_a, Cdim, Cdim);
    k_pack_b_bf16<<<GRID1(CC), T, 0, stream>>>(pos_wq, wq_p, Cdim, Cdim);
    k_pack_b_bf16<<<GRID1(CC), T, 0, stream>>>(pos_wk, wk_p, Cdim, Cdim);
    k_pack_b_bf16<<<GRID1(Cdim * HIDdim), T, 0, stream>>>(ff_w1, w1_b, Cdim, HIDdim);
    k_pack_b_bf16<<<GRID1(HIDdim * Cdim), T, 0, stream>>>(ff_w2, w2_b, HIDdim, Cdim);

    const dim3 gemmBlk(128);                         // 4 waves, 16x64 tile each
    const dim3 gridN256(Lnodes / 16, Cdim / 256);    // (1250, 1)
    const dim3 gridN512(Lnodes / 16, HIDdim / 256);  // (1250, 2)
    const dim3 lnGrid(Lnodes / 8);
    const float NEG_INF = -INFINITY;

    // ================= Block 1: sparse attention =================
    k_layernorm<<<lnGrid, T, 0, stream>>>(x_in, ln_att_g, ln_att_b, z, xcur);
    k_gemm_wmma<Cdim, false><<<gridN256, gemmBlk, 0, stream>>>(z, nullptr, wq_a, att_bq, nullptr, nullptr, q, Cdim, 0);
    k_gemm_wmma<Cdim, false><<<gridN256, gemmBlk, 0, stream>>>(z, nullptr, wk_a, att_bk, nullptr, nullptr, k, Cdim, 0);
    k_gemm_wmma<Cdim, false><<<gridN256, gemmBlk, 0, stream>>>(z, nullptr, wv_a, att_bv, nullptr, nullptr, v, Cdim, 0);
    k_fill_f32<<<GRID1(LH), T, 0, stream>>>(rowmax, NEG_INF, (int)LH);
    k_fill_f32<<<GRID1(LH), T, 0, stream>>>(rowsum, 0.f, (int)LH);
    k_fill_f32<<<GRID1(LC), T, 0, stream>>>(agg, 0.f, (int)LC);
    k_edge_scores<<<GRID1(EH), T, 0, stream>>>(q, k, row_idx, col_idx, att_bias, scores, rowmax);
    k_edge_expsum<<<GRID1(EH), T, 0, stream>>>(scores, row_idx, rowmax, rowsum);
    k_edge_msg<<<GRID1(EH), T, 0, stream>>>(scores, row_idx, col_idx, rowsum, v, agg);
    // xcur += agg @ Wo + bo
    k_gemm_wmma<Cdim, false><<<gridN256, gemmBlk, 0, stream>>>(agg, nullptr, wo_a, att_bo, xcur, nullptr, xcur, Cdim, 0);

    // ================= Block 2: positional attention =================
    k_layernorm<<<lnGrid, T, 0, stream>>>(xcur, ln_pos_g, ln_pos_b, z, nullptr);
    k_gemm_wmma<Cdim, false><<<gridN256, gemmBlk, 0, stream>>>(z, nullptr, wq_p, nullptr, nullptr, nullptr, q, Cdim, 0);
    k_gemm_wmma<Cdim, true ><<<gridN256, gemmBlk, 0, stream>>>(z, o2s,     wk_p, nullptr, nullptr, nullptr, k, Cdim, 0);
    k_fill_f32<<<GRID1(LH), T, 0, stream>>>(rowmax, NEG_INF, (int)LH);
    k_fill_f32<<<GRID1(LH), T, 0, stream>>>(rowsum, 0.f, (int)LH);
    k_fill_f32<<<GRID1(Lnodes * 3 * Hheads), T, 0, stream>>>(agg, 0.f, Lnodes * 3 * Hheads);
    k_edge_scores<<<GRID1(EH), T, 0, stream>>>(q, k, row_idx, src_idx, pos_bias, scores, rowmax);
    k_edge_expsum<<<GRID1(EH), T, 0, stream>>>(scores, row_idx, rowmax, rowsum);
    k_pos_msg<<<GRID1(EH), T, 0, stream>>>(scores, row_idx, src_idx, rowsum, dist, pos, src_pos, agg);
    k_pos_wo<<<GRID1(LC), T, 0, stream>>>(agg, pos_wo, pos_bo, xcur);

    // ================= Block 3: FFN (double residual per reference) ==========
    k_layernorm<<<lnGrid, T, 0, stream>>>(xcur, ln_ff_g, ln_ff_b, z, nullptr);
    k_gemm_wmma<Cdim,  false><<<gridN512, gemmBlk, 0, stream>>>(z, nullptr, w1_b, ff_b1, nullptr, nullptr, hidden, HIDdim, 1);
    k_gemm_wmma<HIDdim,false><<<gridN256, gemmBlk, 0, stream>>>(hidden, nullptr, w2_b, ff_b2, z, xcur, (float*)d_out, Cdim, 0);
}
